// ScaledDotProductModule_26448408609121
// MI455X (gfx1250) — compile-verified, matrix operand order checked by reading, not run on hardware
//
#include <hip/hip_runtime.h>
#include <math.h>

// ---------------------------------------------------------------------------
// Problem constants (from reference): B=4, S=2048, D_MODEL=1024, 3*D = 3072
// ---------------------------------------------------------------------------
#define BATCH   4
#define SEQ     2048
#define DM      1024
#define NQKV    3072
#define MTOT    (BATCH * SEQ)        // 8192
#define MASK_NEG (-9.0e15f)
#define SCALE    (0.03125f)          // 1/sqrt(1024)

typedef unsigned int       u32;
typedef unsigned short     u16;
typedef __attribute__((ext_vector_type(16))) __bf16 v16bf;
typedef __attribute__((ext_vector_type(8)))  float  v8f;
typedef __attribute__((ext_vector_type(4)))  u32    u32x4;

union Frag {
    v16bf v;
    u32x4 q[2];
};

// fp32 -> bf16 round-to-nearest-even
static __device__ __forceinline__ u16 f2bf(float f) {
    u32 u = __float_as_uint(f);
    u32 r = u + 0x7FFFu + ((u >> 16) & 1u);
    return (u16)(r >> 16);
}
static __device__ __forceinline__ float bf2f(u16 h) {
    return __uint_as_float(((u32)h) << 16);
}

// Load one 16-bit A/B WMMA fragment (16x32 bf16) from a row-major row pointer.
// Per CDNA5 ISA 7.12.2: lanes 0-15 hold K = k..k+7 and k+16..k+23,
// lanes 16-31 hold K = k+8..k+15 and k+24..k+31  -> two contiguous 16B chunks.
static __device__ __forceinline__ void load_frag(Frag& f, const u16* rowk, int hi) {
    const u16* p = rowk + hi * 8;
    f.q[0] = *(const u32x4*)(p);
    f.q[1] = *(const u32x4*)(p + 16);
}

static __device__ __forceinline__ v8f wmma_bf16(const Frag& a, const Frag& b, v8f c) {
    // D = A(16x32 bf16) * B(32x16 bf16) + C(16x16 f32)
    return __builtin_amdgcn_wmma_f32_16x16x32_bf16(
        /*neg_a=*/false, a.v, /*neg_b=*/false, b.v,
        /*c_mod=*/(short)0, c, /*reuse_a=*/false, /*reuse_b=*/false);
}

// ---------------------------------------------------------------------------
// Kernel 1: fp32 x -> bf16 (row-major [8192][1024])
// ---------------------------------------------------------------------------
__global__ __launch_bounds__(256) void cvt_x_kernel(const float* __restrict__ x,
                                                    u16* __restrict__ xb) {
    int i = blockIdx.x * 256 + threadIdx.x;   // over 8388608 elems
    xb[i] = f2bf(x[i]);
}

// ---------------------------------------------------------------------------
// Kernel 2: fp32 W[1024][3072] -> bf16 transposed Wt[3072][1024]
// ---------------------------------------------------------------------------
__global__ __launch_bounds__(256) void cvt_wt_kernel(const float* __restrict__ W,
                                                     u16* __restrict__ wt) {
    int i = blockIdx.x * 256 + threadIdx.x;   // over 3145728 elems
    int n = i >> 10;                          // output row (0..3071)
    int k = i & 1023;                         // output col (0..1023)
    wt[i] = f2bf(W[k * NQKV + n]);
}

// ---------------------------------------------------------------------------
// Kernel 3: QKV projection, bf16 WMMA, f32 accumulate.
//   C[8192][3072] = Xbf[8192][1024] @ Wt^T   (Wt stored [3072][1024])
// Output routed: cols [0,1024)->Q[m][d], [1024,2048)->K[m][d],
//                [2048,3072)->Vt[b][d][s] (transposed for the P@V matmul).
// Block = 256 thr (8 waves), block tile 128x128, wave tile 64x32.
// Operands are L2-resident (X 16MB + Wt 6MB << 192MB L2): no LDS staging.
// ---------------------------------------------------------------------------
__global__ __launch_bounds__(256) void qkv_gemm_kernel(const u16* __restrict__ Xb,
                                                       const u16* __restrict__ Wt,
                                                       u16* __restrict__ Qb,
                                                       u16* __restrict__ Kb,
                                                       u16* __restrict__ Vt) {
    const int lane = threadIdx.x & 31;
    const int wave = threadIdx.x >> 5;
    const int l16  = lane & 15;
    const int hi   = lane >> 4;

    const int mBase = blockIdx.y * 128 + (wave >> 2) * 64;   // 4 M tiles
    const int nBase = blockIdx.x * 128 + (wave & 3) * 32;    // 2 N tiles

    v8f acc[4][2];
#pragma unroll
    for (int i = 0; i < 4; ++i)
#pragma unroll
        for (int j = 0; j < 2; ++j) { v8f z = {}; acc[i][j] = z; }

    const u16* aRow[4];
    const u16* bRow[2];
#pragma unroll
    for (int i = 0; i < 4; ++i) aRow[i] = Xb + (size_t)(mBase + i * 16 + l16) * DM;
#pragma unroll
    for (int j = 0; j < 2; ++j) bRow[j] = Wt + (size_t)(nBase + j * 16 + l16) * DM;

    for (int kd = 0; kd < DM; kd += 32) {
        Frag a[4], b[2];
#pragma unroll
        for (int i = 0; i < 4; ++i) load_frag(a[i], aRow[i] + kd, hi);
#pragma unroll
        for (int j = 0; j < 2; ++j) load_frag(b[j], bRow[j] + kd, hi);
        if (kd + 64 < DM) {   // WGP-scope prefetch (locality=3) of next-next chunk
            __builtin_prefetch(aRow[0] + kd + 64, 0, 3);
            __builtin_prefetch(bRow[0] + kd + 64, 0, 3);
        }
#pragma unroll
        for (int i = 0; i < 4; ++i)
#pragma unroll
            for (int j = 0; j < 2; ++j)
                acc[i][j] = wmma_bf16(a[i], b[j], acc[i][j]);
    }

    // Store accumulators. C/D layout: lane holds N = l16, rows M = r + 8*hi.
#pragma unroll
    for (int i = 0; i < 4; ++i) {
#pragma unroll
        for (int j = 0; j < 2; ++j) {
            const int n0     = nBase + j * 16;
            const int stream = n0 >> 10;            // uniform per tile
            const int col    = (n0 & 1023) + l16;   // 0..1023 within stream
#pragma unroll
            for (int r = 0; r < 8; ++r) {
                const int m  = mBase + i * 16 + r + 8 * hi;   // global row 0..8191
                const u16 bv = f2bf(acc[i][j][r]);
                if (stream == 0) {
                    Qb[(size_t)m * DM + col] = bv;
                } else if (stream == 1) {
                    Kb[(size_t)m * DM + col] = bv;
                } else {
                    const int bb = m >> 11;         // batch
                    const int s  = m & 2047;        // seq pos
                    Vt[(size_t)bb * (DM * SEQ) + (size_t)col * SEQ + s] = bv;
                }
            }
        }
    }
}

// ---------------------------------------------------------------------------
// Kernel 4: fused masked attention for one (batch, 16-query tile).
//   Phase 1: S = (Q Kt)/32 with triu mask -> bf16 scores in LDS [16][2048]
//            4-way key-tile register blocking: 1 Q fragment feeds 4 WMMAs
//   Phase 2: f32 row softmax in LDS (bf16 storage, f32 math)
//   Phase 3: O = P @ V via WMMA, A-fragments from LDS, 4-way dim blocking
// Block = 256 thr (8 waves). Grid = 4 * 128 = 512 blocks.
// ---------------------------------------------------------------------------
__global__ __launch_bounds__(256) void attn_kernel(const u16* __restrict__ Qb,
                                                   const u16* __restrict__ Kb,
                                                   const u16* __restrict__ Vt,
                                                   float* __restrict__ out) {
    __shared__ u16   sS[16][SEQ];     // 64 KB: scores, then P (bf16)
    __shared__ float sRed[16][16];    // softmax reduction scratch

    const int tid  = threadIdx.x;
    const int lane = tid & 31;
    const int wave = tid >> 5;
    const int l16  = lane & 15;
    const int hi   = lane >> 4;

    const int b  = blockIdx.x >> 7;          // batch
    const int q0 = (blockIdx.x & 127) << 4;  // first query of tile

    const u16 negbf = f2bf(MASK_NEG);

    // ---- Phase 1: score strips (each wave: 256 keys = 4 groups x 4 tiles) --
    const u16* Qrow = Qb + (size_t)(b * SEQ + q0 + l16) * DM;  // lane = query row
    for (int ng = 0; ng < 4; ++ng) {
        const int jG = wave * 256 + ng * 64;     // group of 64 keys
        if (jG + 64 <= q0) {
            // whole group below diagonal: masked, skip the matmuls entirely
#pragma unroll
            for (int t = 0; t < 4; ++t)
#pragma unroll
                for (int r = 0; r < 8; ++r)
                    sS[r + 8 * hi][jG + t * 16 + l16] = negbf;
            continue;
        }
        v8f acc[4];
#pragma unroll
        for (int t = 0; t < 4; ++t) { v8f z = {}; acc[t] = z; }
        const u16* Krow[4];
#pragma unroll
        for (int t = 0; t < 4; ++t)
            Krow[t] = Kb + (size_t)(b * SEQ + jG + t * 16 + l16) * DM;
        for (int kd = 0; kd < DM; kd += 32) {
            Frag a;
            load_frag(a, Qrow + kd, hi);         // one A fragment -> 4 WMMAs
            Frag bb[4];
#pragma unroll
            for (int t = 0; t < 4; ++t) load_frag(bb[t], Krow[t] + kd, hi);
#pragma unroll
            for (int t = 0; t < 4; ++t) acc[t] = wmma_bf16(a, bb[t], acc[t]);
        }
#pragma unroll
        for (int t = 0; t < 4; ++t) {
            const int j = jG + t * 16 + l16;     // this lane's key column
#pragma unroll
            for (int r = 0; r < 8; ++r) {
                const int m   = r + 8 * hi;      // query row in tile
                const int qi  = q0 + m;          // global query index
                const float v = acc[t][r] * SCALE;
                // faithful to ref: triu keeps j>=i, then where(score==0) -> MASK_NEG
                const bool keep = (j >= qi) && (v != 0.0f);
                sS[m][j] = keep ? f2bf(v) : negbf;
            }
        }
    }
    __syncthreads();

    // ---- Phase 2: row softmax (16 rows x 2048; 16 threads per row) ---------
    {
        const int row = tid >> 4;
        const int c0  = tid & 15;
        float mx = -3.0e38f;
        for (int i = 0; i < 128; ++i)
            mx = fmaxf(mx, bf2f(sS[row][c0 + (i << 4)]));
        sRed[row][c0] = mx;
        __syncthreads();
        float rmax = sRed[row][0];
#pragma unroll
        for (int i = 1; i < 16; ++i) rmax = fmaxf(rmax, sRed[row][i]);
        __syncthreads();                      // before sRed reuse
        float sum = 0.0f;
        for (int i = 0; i < 128; ++i) {
            const int j   = c0 + (i << 4);
            const float e = __expf(bf2f(sS[row][j]) - rmax);
            sum += e;
            sS[row][j] = f2bf(e);             // unnormalized p
        }
        sRed[row][c0] = sum;
        __syncthreads();
        float tot = 0.0f;
#pragma unroll
        for (int i = 0; i < 16; ++i) tot += sRed[row][i];
        const float inv = 1.0f / tot;
        for (int i = 0; i < 128; ++i) {
            const int j = c0 + (i << 4);
            sS[row][j] = f2bf(bf2f(sS[row][j]) * inv);
        }
    }
    __syncthreads();

    // ---- Phase 3: O = P @ V (each wave: 128 dims = 2 groups x 4 tiles) -----
    const int  kStart = q0 & ~31;             // keys < q0 have P == 0
    const u16* VtB    = Vt + (size_t)b * (DM * SEQ);
    const u16* Prow   = &sS[l16][0];          // lane = query row (A fragment)
    for (int dg = 0; dg < 2; ++dg) {
        const int dG = wave * 128 + dg * 64;
        v8f acc[4];
#pragma unroll
        for (int t = 0; t < 4; ++t) { v8f z = {}; acc[t] = z; }
        const u16* Vrow[4];
#pragma unroll
        for (int t = 0; t < 4; ++t)
            Vrow[t] = VtB + (size_t)(dG + t * 16 + l16) * SEQ;   // lane = dim row
        for (int kk = kStart; kk < SEQ; kk += 32) {
            Frag a;
            load_frag(a, Prow + kk, hi);      // ds_load_b128 from LDS, feeds 4 WMMAs
            Frag bb[4];
#pragma unroll
            for (int t = 0; t < 4; ++t) load_frag(bb[t], Vrow[t] + kk, hi);
#pragma unroll
            for (int t = 0; t < 4; ++t) acc[t] = wmma_bf16(a, bb[t], acc[t]);
        }
#pragma unroll
        for (int t = 0; t < 4; ++t) {
            const int d = dG + t * 16 + l16;
#pragma unroll
            for (int r = 0; r < 8; ++r) {
                const int m = r + 8 * hi;
                out[(size_t)(b * SEQ + q0 + m) * DM + d] = acc[t][r];
            }
        }
    }
}

// ---------------------------------------------------------------------------
// Host-side launcher
// ---------------------------------------------------------------------------
extern "C" void kernel_launch(void* const* d_in, const int* in_sizes, int n_in,
                              void* d_out, int out_size, void* d_ws, size_t ws_size,
                              hipStream_t stream) {
    const float* x = (const float*)d_in[0];   // [4][2048][1024] f32
    const float* W = (const float*)d_in[1];   // [1024][3072] f32
    float* out = (float*)d_out;               // [4][2048][1024] f32

    // Workspace carve-up (bf16 = u16), total ~70 MB
    u16* Xb = (u16*)d_ws;                     // [8192][1024]
    u16* Wt = Xb + (size_t)MTOT * DM;         // [3072][1024]  (W transposed)
    u16* Qb = Wt + (size_t)NQKV * DM;         // [8192][1024]
    u16* Kb = Qb + (size_t)MTOT * DM;         // [8192][1024]
    u16* Vt = Kb + (size_t)MTOT * DM;         // [4][1024][2048] (v transposed)

    cvt_x_kernel<<<(MTOT * DM) / 256, 256, 0, stream>>>(x, Xb);
    cvt_wt_kernel<<<(NQKV * DM) / 256, 256, 0, stream>>>(W, Wt);

    qkv_gemm_kernel<<<dim3(NQKV / 128, MTOT / 128), 256, 0, stream>>>(Xb, Wt, Qb, Kb, Vt);

    attn_kernel<<<BATCH * (SEQ / 16), 256, 0, stream>>>(Qb, Kb, Vt, out);
}
